// ColAttenPairBias_23081154248862
// MI455X (gfx1250) — compile-verified, matrix operand order checked by reading, not run on hardware
//
#include <hip/hip_runtime.h>
#include <hip/hip_bf16.h>

// ---------------------------------------------------------------------------
// ColAttenPairBias for MI455X (gfx1250): bf16 WMMA attention, fused per-column.
// B=1, d_m=256, n=128, L=1024, H=8, dh=32, p_d=128.
// - All WMMA B-operands stored N-major so A/B fragments load as 2x ds_load_b128.
// - Wave-uniform tile indices scalarized via readfirstlane.
// - Global->LDS staging uses CDNA5 async-to-LDS ops (ASYNCcnt).
// - Native f32->bf16 conversion (v_cvt) instead of manual bit-twiddling.
// ---------------------------------------------------------------------------

typedef __bf16 bf16;
typedef __attribute__((ext_vector_type(16))) __bf16 v16bf;
typedef __attribute__((ext_vector_type(8)))  float  v8f;

#define D_M 256
#define NSEQ 128
#define LLEN 1024
#define NHEAD 8
#define DH 32
#define PD 128

__device__ __forceinline__ bf16 f2bf(float f) { return (bf16)f; }

// Low 32 bits of a generic pointer to __shared__ = LDS byte offset (ISA 10.2).
__device__ __forceinline__ unsigned lds_off(const void* p) {
  return (unsigned)(unsigned long long)p;
}

// CDNA5 async global->LDS copy, 16 B per lane (ISA 15.18.3 op 98, ASYNCcnt).
__device__ __forceinline__ void async_g2l_b128(unsigned lds, const void* g) {
  asm volatile("global_load_async_to_lds_b128 %0, %1, off"
               :: "v"(lds), "v"(g) : "memory");
}
__device__ __forceinline__ void wait_async0() {
  asm volatile("s_wait_asynccnt 0" ::: "memory");
}

// Fragment loader (A and Bt operands). Source: [16 rows][K] row-major, leading
// dim `ld` (elements). ISA 7.12.2 striping: row = lane&15, K base = (lane>>4)*8;
// h[0..7] -> K=kb..kb+7, h[8..15] -> K=kb+16..kb+23. 2x ds_load_b128 per lane.
__device__ __forceinline__ v16bf ld_frag(const bf16* p, int ld) {
  int lane = threadIdx.x & 31;
  int m  = lane & 15;
  int kb = (lane >> 4) << 3;
  v16bf a;
#pragma unroll
  for (int j = 0; j < 8; ++j) {
    a[j]     = p[m * ld + kb + j];
    a[j + 8] = p[m * ld + kb + 16 + j];
  }
  return a;
}

#define WMMA_BF16(a, b, c) \
  __builtin_amdgcn_wmma_f32_16x16x32_bf16(false, (a), false, (b), (short)0, (c), false, false)

// ---------------------------------------------------------------------------
// Kernel 1: f32 -> bf16 with transpose: wT[o][i] = W[i][o]   (256x256)
// ---------------------------------------------------------------------------
__global__ __launch_bounds__(256) void cvt_transpose_kernel(
    const float* __restrict__ W, bf16* __restrict__ wT) {
  int idx = blockIdx.x * 256 + threadIdx.x;   // = o*256 + i
  int o = idx >> 8;
  int i = idx & 255;
  wT[idx] = f2bf(W[i * D_M + o]);
}

// ---------------------------------------------------------------------------
// Kernel 2: LayerNorm over d_m + transpose msa[d_m,n,L] -> x_bf16[L,n,d_m]
// ---------------------------------------------------------------------------
__global__ __launch_bounds__(256) void ln_transpose_kernel(
    const float* __restrict__ msa, const float* __restrict__ g,
    const float* __restrict__ b, bf16* __restrict__ xbf) {
  int i = blockIdx.x;
  int l = blockIdx.y * 256 + threadIdx.x;
  const float* base = msa + (size_t)i * LLEN + l;  // + c * (n*L)
  float s = 0.f, ss = 0.f;
#pragma unroll 4
  for (int c = 0; c < D_M; ++c) {
    float x = base[(size_t)c * (NSEQ * LLEN)];
    s += x; ss += x * x;
  }
  float mean = s * (1.f / D_M);
  float var  = ss * (1.f / D_M) - mean * mean;
  float rstd = rsqrtf(var + 1e-5f);
  bf16* dst = xbf + (size_t)l * (NSEQ * D_M) + (size_t)i * D_M;
  union { bf16 h[8]; uint4 v; } buf;
  for (int c0 = 0; c0 < D_M; c0 += 8) {
#pragma unroll
    for (int j = 0; j < 8; ++j) {
      float x = base[(size_t)(c0 + j) * (NSEQ * LLEN)];
      buf.h[j] = f2bf((x - mean) * rstd * g[c0 + j] + b[c0 + j]);
    }
    *(uint4*)(dst + c0) = buf.v;
  }
}

// ---------------------------------------------------------------------------
// Kernel 3: pair LN + bias = LN(pairs) @ Wb -> bias[H][n][n] f32
// ---------------------------------------------------------------------------
__global__ __launch_bounds__(128) void pair_bias_kernel(
    const float* __restrict__ pairs, const float* __restrict__ g,
    const float* __restrict__ b, const float* __restrict__ Wb,
    float* __restrict__ bias) {
  int i = blockIdx.x, j = threadIdx.x;
  const float* row = pairs + ((size_t)i * NSEQ + j) * PD;
  float s = 0.f, ss = 0.f;
  for (int c = 0; c < PD; ++c) { float x = row[c]; s += x; ss += x * x; }
  float mean = s * (1.f / PD);
  float var  = ss * (1.f / PD) - mean * mean;
  float rstd = rsqrtf(var + 1e-5f);
  float acc[NHEAD] = {};
  for (int c = 0; c < PD; ++c) {
    float xn = (row[c] - mean) * rstd * g[c] + b[c];
#pragma unroll
    for (int h = 0; h < NHEAD; ++h) acc[h] += xn * Wb[c * NHEAD + h];
  }
  for (int h = 0; h < NHEAD; ++h)
    bias[(size_t)h * (NSEQ * NSEQ) + (size_t)i * NSEQ + j] = acc[h];
}

// ---------------------------------------------------------------------------
// Kernel 4: fused per-column attention. grid = L (1024), block = 256 (8 waves).
// ---------------------------------------------------------------------------
__global__ __launch_bounds__(256, 1) void col_attn_kernel(
    const bf16* __restrict__ xbf,
    const bf16* __restrict__ wqT, const bf16* __restrict__ wkT,
    const bf16* __restrict__ wvT, const bf16* __restrict__ wgT,
    const bf16* __restrict__ woT,
    const float* __restrict__ bias,   // [H][128][128]
    const float* __restrict__ bg, const float* __restrict__ bo,
    float* __restrict__ out) {
  __shared__ __align__(16) bf16  sX[NSEQ * D_M];      // 64 KB, LN'd inputs [n][d_m]
  __shared__ __align__(16) bf16  sAttn[NSEQ * D_M];   // 64 KB, gated attn [n][d_m]
  __shared__ __align__(16) float sS[NSEQ * NSEQ];     // 64 KB, scores (aliased: sW/sWoT)
  __shared__ __align__(16) bf16  sP[NSEQ * NSEQ];     // 32 KB, probs [n][n]
  __shared__ __align__(16) bf16  sQ[NSEQ * DH];       // 8 KB  [n][dh]
  __shared__ __align__(16) bf16  sKh[NSEQ * DH];      // 8 KB  [n][dh]
  __shared__ __align__(16) bf16  sVT[DH * NSEQ];      // 8 KB  [dh][n]
  __shared__ __align__(16) float sG[NSEQ * DH];       // 16 KB, sigmoid gate

  const int l    = blockIdx.x;
  const int tid  = threadIdx.x;
  const int lane = tid & 31;
  const int swid = __builtin_amdgcn_readfirstlane(tid >> 5);  // scalar wave id
  const int fn   = lane & 15;          // fragment column
  const int fmb  = (lane >> 4) << 3;   // fragment row base

  // ---- stage X[l] : 128x256 bf16 = 4096 x 16 B, async to LDS ----
  {
    const uint4* src = (const uint4*)(xbf + (size_t)l * (NSEQ * D_M));
    unsigned dst = lds_off(sX);
#pragma unroll
    for (int it = 0; it < 16; ++it) {
      int idx = tid + it * 256;
      async_g2l_b128(dst + idx * 16, src + idx);
    }
    wait_async0();
  }
  __syncthreads();

  const bf16* wmats[4] = {wqT, wkT, wvT, wgT};
  const float scale = 0.1767766953f;  // 1/sqrt(32)

  for (int h = 0; h < NHEAD; ++h) {
    // ---- stage per-head transposed weight blocks into sW (aliases sS):
    //      4 x [32 outs][256 ins] bf16 = 64 KB; row r of 128 = (mat, out col).
    bf16* sW = (bf16*)sS;
    {
      int r   = tid >> 1;        // 0..127
      int mat = r >> 5;          // 0..3  (uniform per contiguous thread group)
      int o   = r & 31;
      const uint4* s4 =
          (const uint4*)(wmats[mat] + ((size_t)(h * DH + o)) * D_M) + (tid & 1) * 16;
      unsigned d = lds_off(sW + (mat * DH + o) * D_M) + (tid & 1) * 256;
#pragma unroll
      for (int j = 0; j < 16; ++j) async_g2l_b128(d + j * 16, s4 + j);
      wait_async0();
    }
    __syncthreads();

    // ---- projections Q,K,V,G = X @ W  (64 tiles of 16x16, K-loop of 8)
#pragma unroll 1
    for (int it = 0; it < 8; ++it) {
      int t   = swid + it * 8;                               // scalar
      int mat = __builtin_amdgcn_readfirstlane(t >> 4);
      int rem = t & 15;
      int mt  = __builtin_amdgcn_readfirstlane(rem >> 1);
      int nt  = __builtin_amdgcn_readfirstlane(rem & 1);
      const bf16* wbase = sW + (mat * DH + nt * 16) * D_M;
      v8f acc = {};
#pragma unroll
      for (int kk = 0; kk < 8; ++kk) {
        v16bf a = ld_frag(sX + (mt * 16) * D_M + kk * 32, D_M);
        v16bf b = ld_frag(wbase + kk * 32, D_M);
        acc = WMMA_BF16(a, b, acc);
      }
      if (mat == 0) {
#pragma unroll
        for (int r = 0; r < 8; ++r)
          sQ[(mt * 16 + fmb + r) * DH + nt * 16 + fn] = f2bf(acc[r]);
      } else if (mat == 1) {
#pragma unroll
        for (int r = 0; r < 8; ++r)
          sKh[(mt * 16 + fmb + r) * DH + nt * 16 + fn] = f2bf(acc[r]);
      } else if (mat == 2) {
#pragma unroll
        for (int r = 0; r < 8; ++r)
          sVT[(nt * 16 + fn) * NSEQ + mt * 16 + fmb + r] = f2bf(acc[r]);
      } else {
        float bgv = bg[h * DH + nt * 16 + fn];
#pragma unroll
        for (int r = 0; r < 8; ++r) {
          float z = acc[r] + bgv;
          sG[(mt * 16 + fmb + r) * DH + nt * 16 + fn] = 1.f / (1.f + __expf(-z));
        }
      }
    }
    __syncthreads();

    // ---- scores = Q @ K^T * scale + bias[h]   (8x8 tiles, K=dh=32 -> 1 WMMA)
    const float* biasg = bias + (size_t)h * (NSEQ * NSEQ);
#pragma unroll 1
    for (int it = 0; it < 8; ++it) {
      int t  = swid + it * 8;
      int mt = __builtin_amdgcn_readfirstlane(t >> 3);
      int nt = __builtin_amdgcn_readfirstlane(t & 7);
      v16bf a = ld_frag(sQ + (mt * 16) * DH, DH);
      v16bf b = ld_frag(sKh + (nt * 16) * DH, DH);
      v8f acc = {};
      acc = WMMA_BF16(a, b, acc);
#pragma unroll
      for (int r = 0; r < 8; ++r) {
        int row = mt * 16 + fmb + r;
        int col = nt * 16 + fn;
        sS[row * NSEQ + col] = acc[r] * scale + biasg[row * NSEQ + col];
      }
    }
    __syncthreads();

    // ---- row softmax -> sP (bf16); 2 threads per row, shfl_xor combine
    {
      int r    = tid >> 1;
      int half = tid & 1;
      const float* srow = sS + r * NSEQ + half * 64;
      float m = -3.4e38f;
#pragma unroll 8
      for (int j = 0; j < 64; ++j) m = fmaxf(m, srow[j]);
      m = fmaxf(m, __shfl_xor(m, 1, 32));
      float sum = 0.f;
#pragma unroll 8
      for (int j = 0; j < 64; ++j) sum += __expf(srow[j] - m);
      sum += __shfl_xor(sum, 1, 32);
      float inv = 1.f / sum;
      uint4* prow = (uint4*)(sP + r * NSEQ + half * 64);
      union { bf16 hh[8]; uint4 v; } buf;
      for (int j0 = 0; j0 < 64; j0 += 8) {
#pragma unroll
        for (int jj = 0; jj < 8; ++jj)
          buf.hh[jj] = f2bf(__expf(srow[j0 + jj] - m) * inv);
        prow[j0 >> 3] = buf.v;
      }
    }
    __syncthreads();

    // ---- attn = g * (P @ V) -> sAttn[:, h*32:(h+1)*32]  (16 tiles, K=128)
#pragma unroll 1
    for (int it = 0; it < 2; ++it) {
      int t  = swid + it * 8;
      int mt = __builtin_amdgcn_readfirstlane(t >> 1);
      int nt = __builtin_amdgcn_readfirstlane(t & 1);
      v8f acc = {};
#pragma unroll
      for (int kk = 0; kk < 4; ++kk) {
        v16bf a = ld_frag(sP + (mt * 16) * NSEQ + kk * 32, NSEQ);
        v16bf b = ld_frag(sVT + (nt * 16) * NSEQ + kk * 32, NSEQ);
        acc = WMMA_BF16(a, b, acc);
      }
#pragma unroll
      for (int r = 0; r < 8; ++r) {
        int row = mt * 16 + fmb + r;
        int col = nt * 16 + fn;
        float v = acc[r] * sG[row * DH + col];
        sAttn[row * D_M + h * DH + col] = f2bf(v);
      }
    }
    __syncthreads();
  }

  // ---- out = attn @ Wo + bo, transposed store to out[d_m][n][L]
#pragma unroll 1
  for (int nhalf = 0; nhalf < 2; ++nhalf) {
    bf16* sWoT = (bf16*)sS;  // [128 outs][256 ins] bf16 = 64 KB
    {
      int r = tid >> 1;      // 0..127 : out channel within half
      const uint4* s4 =
          (const uint4*)(woT + ((size_t)(nhalf * 128 + r)) * D_M) + (tid & 1) * 16;
      unsigned d = lds_off(sWoT + r * D_M) + (tid & 1) * 256;
#pragma unroll
      for (int j = 0; j < 16; ++j) async_g2l_b128(d + j * 16, s4 + j);
      wait_async0();
    }
    __syncthreads();
#pragma unroll 1
    for (int it = 0; it < 8; ++it) {
      int t  = swid + it * 8;
      int mt = __builtin_amdgcn_readfirstlane(t >> 3);
      int nt = __builtin_amdgcn_readfirstlane(t & 7);
      v8f acc = {};
#pragma unroll
      for (int kk = 0; kk < 8; ++kk) {
        v16bf a = ld_frag(sAttn + (mt * 16) * D_M + kk * 32, D_M);
        v16bf b = ld_frag(sWoT + (nt * 16) * D_M + kk * 32, D_M);
        acc = WMMA_BF16(a, b, acc);
      }
#pragma unroll
      for (int r = 0; r < 8; ++r) {
        int row  = mt * 16 + fmb + r;             // seq index i
        int colg = nhalf * 128 + nt * 16 + fn;    // channel c
        out[(size_t)colg * (NSEQ * LLEN) + (size_t)row * LLEN + l] =
            acc[r] + bo[colg];
      }
    }
    __syncthreads();
  }
}

// ---------------------------------------------------------------------------
// Host launcher
// ---------------------------------------------------------------------------
extern "C" void kernel_launch(void* const* d_in, const int* in_sizes, int n_in,
                              void* d_out, int out_size, void* d_ws, size_t ws_size,
                              hipStream_t stream) {
  (void)in_sizes; (void)n_in; (void)out_size; (void)ws_size;
  const float* msa    = (const float*)d_in[0];
  const float* pairs  = (const float*)d_in[1];
  const float* ln_m_g = (const float*)d_in[2];
  const float* ln_m_b = (const float*)d_in[3];
  const float* ln_p_g = (const float*)d_in[4];
  const float* ln_p_b = (const float*)d_in[5];
  const float* Wq     = (const float*)d_in[6];
  const float* Wk     = (const float*)d_in[7];
  const float* Wv     = (const float*)d_in[8];
  const float* Wg     = (const float*)d_in[9];
  const float* bg     = (const float*)d_in[10];
  const float* Wb     = (const float*)d_in[11];
  const float* Wo     = (const float*)d_in[12];
  const float* bo     = (const float*)d_in[13];
  float* out = (float*)d_out;

  // workspace layout
  char* ws = (char*)d_ws;
  bf16*  xbf    = (bf16*)ws;                               // 64 MB
  size_t off    = (size_t)LLEN * NSEQ * D_M * sizeof(bf16);
  float* biasws = (float*)(ws + off);                      // 512 KB
  off += (size_t)NHEAD * NSEQ * NSEQ * sizeof(float);
  bf16* wqT = (bf16*)(ws + off); off += (size_t)D_M * D_M * sizeof(bf16);
  bf16* wkT = (bf16*)(ws + off); off += (size_t)D_M * D_M * sizeof(bf16);
  bf16* wvT = (bf16*)(ws + off); off += (size_t)D_M * D_M * sizeof(bf16);
  bf16* wgT = (bf16*)(ws + off); off += (size_t)D_M * D_M * sizeof(bf16);
  bf16* woT = (bf16*)(ws + off); off += (size_t)D_M * D_M * sizeof(bf16);

  const int wel = D_M * D_M;  // 65536
  cvt_transpose_kernel<<<wel / 256, 256, 0, stream>>>(Wq, wqT);
  cvt_transpose_kernel<<<wel / 256, 256, 0, stream>>>(Wk, wkT);
  cvt_transpose_kernel<<<wel / 256, 256, 0, stream>>>(Wv, wvT);
  cvt_transpose_kernel<<<wel / 256, 256, 0, stream>>>(Wg, wgT);
  cvt_transpose_kernel<<<wel / 256, 256, 0, stream>>>(Wo, woT);

  ln_transpose_kernel<<<dim3(NSEQ, LLEN / 256), 256, 0, stream>>>(
      msa, ln_m_g, ln_m_b, xbf);

  pair_bias_kernel<<<NSEQ, NSEQ, 0, stream>>>(pairs, ln_p_g, ln_p_b, Wb, biasws);

  col_attn_kernel<<<LLEN, 256, 0, stream>>>(
      xbf, wqT, wkT, wvT, wgT, woT, biasws, bg, bo, out);
}
